// SelfAttentionBlock_74363063762953
// MI455X (gfx1250) — matrix operand order repaired and checked by reading, not compiled, not run on hardware
//
#include <hip/hip_runtime.h>
#include <hip/hip_bf16.h>
#include <stdint.h>

typedef __attribute__((ext_vector_type(16))) __bf16 v16bf;
typedef __attribute__((ext_vector_type(8)))  float  v8f;
typedef __attribute__((ext_vector_type(4)))  int    v4i;

#define CDIM 256
#define CQK  32
#define NPIX 4096
#define BATCH 4

#define AS1 __attribute__((address_space(1)))
#define AS3 __attribute__((address_space(3)))

// --- CDNA5 async global->LDS load (ASYNCcnt) with inline-asm fallback -------
#if __has_builtin(__builtin_amdgcn_global_load_async_to_lds_b128)
#define ASYNC_LD_B128(gp, lp, off)                                             \
    __builtin_amdgcn_global_load_async_to_lds_b128(                            \
        (AS1 v4i*)(AS1 void*)(gp),                                             \
        (AS3 v4i*)(AS3 void*)(lp), (off), 0)
#else
#define ASYNC_LD_B128(gp, lp, off)                                             \
    asm volatile("global_load_async_to_lds_b128 %0, %1, off offset:" #off     \
                 :: "v"((uint32_t)(uintptr_t)(AS3 void*)(lp)),                 \
                    "v"((AS1 void*)(gp))                                       \
                 : "memory")
#endif

#if __has_builtin(__builtin_amdgcn_s_wait_asynccnt)
#define WAIT_ASYNC(n) __builtin_amdgcn_s_wait_asynccnt(n)
#else
#define WAIT_ASYNC(n) asm volatile("s_wait_asynccnt %0" :: "i"(n) : "memory")
#endif

// round-to-nearest-even float->bf16, packed pair
__device__ __forceinline__ uint32_t pack_bf2(float a, float b) {
    uint32_t ua = __builtin_bit_cast(uint32_t, a);
    uint32_t ub = __builtin_bit_cast(uint32_t, b);
    ua += 0x7FFFu + ((ua >> 16) & 1u);
    ub += 0x7FFFu + ((ub >> 16) & 1u);
    return (ua >> 16) | (ub & 0xFFFF0000u);
}

union U16BF {
    v16bf bf;
    uint32_t u[8];
    uint4 q[2];
};

// ---------------------------------------------------------------------------
// Projection: q[b][n][d], kT[b][n][d] in bf16 (rows = WMMA-friendly layout)
// ---------------------------------------------------------------------------
__global__ __launch_bounds__(256) void proj_qk_kernel(
    const float* __restrict__ x,
    const float* __restrict__ wq, const float* __restrict__ bq,
    const float* __restrict__ wk, const float* __restrict__ bk,
    uint16_t* __restrict__ qbf, uint16_t* __restrict__ ktbf)
{
    __shared__ float wqs[CQK * CDIM];
    __shared__ float wks[CQK * CDIM];
    const int tid = threadIdx.x;
    for (int idx = tid; idx < CQK * CDIM; idx += 256) {
        wqs[idx] = wq[idx];
        wks[idx] = wk[idx];
    }
    __syncthreads();

    const int gn = blockIdx.x * 256 + tid;       // 0 .. B*N-1
    const int b  = gn / NPIX;
    const int n  = gn % NPIX;

    float aq[CQK], ak[CQK];
#pragma unroll
    for (int d = 0; d < CQK; ++d) { aq[d] = bq[d]; ak[d] = bk[d]; }

    const float* xp = x + (size_t)b * CDIM * NPIX + n;
    for (int c = 0; c < CDIM; ++c) {
        float xv = xp[(size_t)c * NPIX];
#pragma unroll
        for (int d = 0; d < CQK; ++d) {
            aq[d] = fmaf(wqs[d * CDIM + c], xv, aq[d]);
            ak[d] = fmaf(wks[d * CDIM + c], xv, ak[d]);
        }
    }
    uint32_t* qrow = (uint32_t*)(qbf  + (size_t)gn * CQK);
    uint32_t* krow = (uint32_t*)(ktbf + (size_t)gn * CQK);
#pragma unroll
    for (int d = 0; d < CQK; d += 2) {
        qrow[d >> 1] = pack_bf2(aq[d], aq[d + 1]);
        krow[d >> 1] = pack_bf2(ak[d], ak[d + 1]);
    }
}

// ---------------------------------------------------------------------------
// Projection: v[b][c][n] in bf16 (j-contiguous -> async tile loads are b128)
// ---------------------------------------------------------------------------
__global__ __launch_bounds__(256) void proj_v_kernel(
    const float* __restrict__ x,
    const float* __restrict__ wv, const float* __restrict__ bv,
    uint16_t* __restrict__ vtbf)
{
    __shared__ float wvs[64 * CDIM];
    const int tid = threadIdx.x;
    const int cq  = blockIdx.y;                  // channel quarter 0..3
    const float* wvp = wv + (size_t)cq * 64 * CDIM;
    for (int idx = tid; idx < 64 * CDIM; idx += 256) wvs[idx] = wvp[idx];
    __syncthreads();

    const int gn = blockIdx.x * 256 + tid;
    const int b  = gn / NPIX;
    const int n  = gn % NPIX;

    float av[64];
#pragma unroll
    for (int d = 0; d < 64; ++d) av[d] = bv[cq * 64 + d];

    const float* xp = x + (size_t)b * CDIM * NPIX + n;
    for (int c = 0; c < CDIM; ++c) {
        float xv = xp[(size_t)c * NPIX];
#pragma unroll
        for (int d = 0; d < 64; ++d) av[d] = fmaf(wvs[d * CDIM + c], xv, av[d]);
    }
    uint16_t* vp = vtbf + ((size_t)b * CDIM + cq * 64) * NPIX + n;
#pragma unroll
    for (int d = 0; d < 64; ++d) {
        uint32_t u = __builtin_bit_cast(uint32_t, av[d]);
        u += 0x7FFFu + ((u >> 16) & 1u);
        vp[(size_t)d * NPIX] = (uint16_t)(u >> 16);
    }
}

// ---------------------------------------------------------------------------
// Flash attention: one wave per 16-query tile, 8 waves/block share V tiles.
//   S^T = K_tile(16xj,32d) x Q^T(32d,16i)  -> softmax stats lane-local
//   V tiles staged via async global->LDS DMA, double-buffered (ASYNCcnt)
//   P transposed via per-wave LDS, O += P(16x32j) x V(32j,16c) per c-block
// ---------------------------------------------------------------------------
#define VPITCH 80   // bytes per V row in LDS (64 data + 16 pad: bank-conflict-free)

__global__ __launch_bounds__(256) void attn_kernel(
    const uint16_t* __restrict__ qbf, const uint16_t* __restrict__ ktbf,
    const uint16_t* __restrict__ vtbf, const float* __restrict__ x,
    const float* __restrict__ gamma, float* __restrict__ out)
{
    __shared__ uint32_t Pbuf[8][16 * 16];                 // per wave: 16 rows x 32 bf16
    __shared__ float    scl[8][16];
    __shared__ __align__(16) uint8_t Vtile[2][CDIM * VPITCH];  // 2 x 20 KB

    const int tid  = threadIdx.x;
    const int wv_  = tid >> 5;
    const int lane = tid & 31;
    const int li   = lane & 15;
    const int half = lane >> 4;

    const int tile = blockIdx.x * 8 + wv_;       // 1024 tiles
    const int b    = tile >> 8;                  // 256 tiles per batch (uniform per block)
    const int i0   = (tile & 255) * 16;

    const uint16_t* kbase = ktbf + (size_t)b * NPIX * CQK;
    const uint16_t* vbase = vtbf + (size_t)b * CDIM * NPIX;

    // Prologue: stage V tile for j0=0 into buffer 0 (thread tid owns row c=tid;
    // async INST_OFFSET applies to both global and LDS addresses).
    {
        const uint16_t* gp = vbase + (size_t)tid * NPIX;
        uint8_t* lp = &Vtile[0][tid * VPITCH];
        ASYNC_LD_B128(gp, lp, 0);
        ASYNC_LD_B128(gp, lp, 16);
        ASYNC_LD_B128(gp, lp, 32);
        ASYNC_LD_B128(gp, lp, 48);
    }

    // Q as B-matrix: lane column i = li; e=0..7 -> d=half*8+e; e=8..15 -> d=16+half*8+(e-8)
    U16BF Qr;
    {
        const uint4* qp = (const uint4*)(qbf + ((size_t)b * NPIX + i0 + li) * CQK + half * 8);
        Qr.q[0] = qp[0];
        Qr.q[1] = qp[2];
    }

    // ---------------- pass 1: streaming row max / sumexp ----------------
    float m = -1e30f, l = 0.0f;
    for (int j0 = 0; j0 < NPIX; j0 += 16) {
        U16BF Kr;  // A-matrix: row j = li; e=0..7 -> d=half*8+e; e=8..15 -> d=16+half*8+(e-8)
        const uint16_t* kr = kbase + (size_t)(j0 + li) * CQK;
        *(uint4*)&Kr.u[0] = *(const uint4*)(kr + half * 8);
        *(uint4*)&Kr.u[4] = *(const uint4*)(kr + 16 + half * 8);
        __builtin_prefetch(kr + 16 * CQK, 0, 1);

        v8f d = {};
        d = __builtin_amdgcn_wmma_f32_16x16x32_bf16(false, Kr.bf, false, Qr.bf,
                                                    (short)0, d, false, false);
        float tm = d[0];
#pragma unroll
        for (int v = 1; v < 8; ++v) tm = fmaxf(tm, d[v]);
        // merge across the half-pair (both lanes hold query i = li)
        float tmo = __builtin_bit_cast(float,
            __builtin_amdgcn_ds_swizzle(__builtin_bit_cast(int, tm), 0x401F)); // SWAPX16
        tm = fmaxf(tm, tmo);
        float nm    = fmaxf(m, tm);
        float alpha = __expf(m - nm);
        float s = 0.0f;
#pragma unroll
        for (int v = 0; v < 8; ++v) s += __expf(d[v] - nm);
        l = l * alpha + s;
        m = nm;
    }
    float lt = l + __builtin_bit_cast(float,
        __builtin_amdgcn_ds_swizzle(__builtin_bit_cast(int, l), 0x401F));
    const float g = gamma[0];
    if (lane < 16) scl[wv_][li] = g / lt;
    asm volatile("" ::: "memory");

    // ---------------- pass 2: recompute S, accumulate O = P x V ----------------
    v8f acc[16];
#pragma unroll
    for (int cb = 0; cb < 16; ++cb) acc[cb] = (v8f){0, 0, 0, 0, 0, 0, 0, 0};

    uint32_t* prow = &Pbuf[wv_][0];
    for (int it = 0; it < NPIX / 32; ++it) {
        const int j0 = it * 32;

        // All waves finished reading the buffer we are about to overwrite.
        __syncthreads();
        {   // stage next V tile (wraps harmlessly on the last iteration)
            const int jn = (j0 + 32) & (NPIX - 1);
            const uint16_t* gp = vbase + (size_t)tid * NPIX + jn;
            uint8_t* lp = &Vtile[(it + 1) & 1][tid * VPITCH];
            ASYNC_LD_B128(gp, lp, 0);
            ASYNC_LD_B128(gp, lp, 16);
            ASYNC_LD_B128(gp, lp, 32);
            ASYNC_LD_B128(gp, lp, 48);
        }
        WAIT_ASYNC(4);        // async loads complete in order: current tile resident
        __syncthreads();      // publish to all waves
        const uint8_t* vbuf = &Vtile[it & 1][0];

        // Both S^T subtiles back-to-back (2nd WMMA fills 1st's hazard slots)
        U16BF Kr0, Kr1;
        {
            const uint16_t* kr0 = kbase + (size_t)(j0 + li) * CQK;
            const uint16_t* kr1 = kbase + (size_t)(j0 + 16 + li) * CQK;
            *(uint4*)&Kr0.u[0] = *(const uint4*)(kr0 + half * 8);
            *(uint4*)&Kr0.u[4] = *(const uint4*)(kr0 + 16 + half * 8);
            *(uint4*)&Kr1.u[0] = *(const uint4*)(kr1 + half * 8);
            *(uint4*)&Kr1.u[4] = *(const uint4*)(kr1 + 16 + half * 8);
        }
        v8f d0 = {}, d1 = {};
        d0 = __builtin_amdgcn_wmma_f32_16x16x32_bf16(false, Kr0.bf, false, Qr.bf,
                                                     (short)0, d0, false, false);
        d1 = __builtin_amdgcn_wmma_f32_16x16x32_bf16(false, Kr1.bf, false, Qr.bf,
                                                     (short)0, d1, false, false);
        // lane holds j = j0 + t*16 + 8*half + v (contiguous), query i = li
        uint4 pk0, pk1;
        pk0.x = pack_bf2(__expf(d0[0] - m), __expf(d0[1] - m));
        pk0.y = pack_bf2(__expf(d0[2] - m), __expf(d0[3] - m));
        pk0.z = pack_bf2(__expf(d0[4] - m), __expf(d0[5] - m));
        pk0.w = pack_bf2(__expf(d0[6] - m), __expf(d0[7] - m));
        pk1.x = pack_bf2(__expf(d1[0] - m), __expf(d1[1] - m));
        pk1.y = pack_bf2(__expf(d1[2] - m), __expf(d1[3] - m));
        pk1.z = pack_bf2(__expf(d1[4] - m), __expf(d1[5] - m));
        pk1.w = pack_bf2(__expf(d1[6] - m), __expf(d1[7] - m));
        *(uint4*)&prow[li * 16 + half * 4]     = pk0;   // cols  0+half*8 .. +7
        *(uint4*)&prow[li * 16 + 8 + half * 4] = pk1;   // cols 16+half*8 .. +7
        asm volatile("" ::: "memory");  // DS ops are in-order per wave

        U16BF Pa;  // A-matrix: row i = li; e=0..7 -> j=half*8+e; e=8..15 -> j=16+half*8+(e-8)
        *(uint4*)&Pa.u[0] = *(uint4*)&prow[li * 16 + half * 4];
        *(uint4*)&Pa.u[4] = *(uint4*)&prow[li * 16 + 8 + half * 4];
        asm volatile("" ::: "memory");

#pragma unroll
        for (int cb = 0; cb < 16; ++cb) {
            U16BF Vr;  // B-matrix: col c = cb*16+li; e=0..7 -> j=half*8+e; e=8..15 -> j=16+half*8+(e-8)
            const uint8_t* vrow = vbuf + (size_t)(cb * 16 + li) * VPITCH + half * 16;
            *(uint4*)&Vr.u[0] = *(const uint4*)(vrow);
            *(uint4*)&Vr.u[4] = *(const uint4*)(vrow + 32);
            acc[cb] = __builtin_amdgcn_wmma_f32_16x16x32_bf16(false, Pa.bf, false, Vr.bf,
                                                              (short)0, acc[cb], false, false);
        }
    }

    // ---------------- epilogue: out = gamma*O/l + x ----------------
    float s0[8];
#pragma unroll
    for (int v = 0; v < 8; ++v) s0[v] = scl[wv_][half * 8 + v];  // scale for row i = v+8*half

#pragma unroll
    for (int cb = 0; cb < 16; ++cb) {
        const int c = cb * 16 + li;
        const size_t base = ((size_t)b * CDIM + c) * NPIX + i0 + half * 8;
        const float4* xr = (const float4*)(x + base);
        float4* orow = (float4*)(out + base);
        float4 x0 = xr[0], x1 = xr[1];
        float4 o0, o1;
        o0.x = acc[cb][0] * s0[0] + x0.x;
        o0.y = acc[cb][1] * s0[1] + x0.y;
        o0.z = acc[cb][2] * s0[2] + x0.z;
        o0.w = acc[cb][3] * s0[3] + x0.w;
        o1.x = acc[cb][4] * s0[4] + x1.x;
        o1.y = acc[cb][5] * s0[5] + x1.y;
        o1.z = acc[cb][6] * s0[6] + x1.z;
        o1.w = acc[cb][7] * s0[7] + x1.w;
        orow[0] = o0;
        orow[1] = o1;
    }
}

// ---------------------------------------------------------------------------
extern "C" void kernel_launch(void* const* d_in, const int* in_sizes, int n_in,
                              void* d_out, int out_size, void* d_ws, size_t ws_size,
                              hipStream_t stream) {
    const float* x     = (const float*)d_in[0];
    const float* wq    = (const float*)d_in[1];
    const float* bq    = (const float*)d_in[2];
    const float* wk    = (const float*)d_in[3];
    const float* bk    = (const float*)d_in[4];
    const float* wv    = (const float*)d_in[5];
    const float* bv    = (const float*)d_in[6];
    const float* gamma = (const float*)d_in[7];
    float* out = (float*)d_out;

    uint16_t* qbf  = (uint16_t*)d_ws;                          // B*N*32 bf16 = 1 MB
    uint16_t* ktbf = qbf  + (size_t)BATCH * NPIX * CQK;        // 1 MB
    uint16_t* vtbf = ktbf + (size_t)BATCH * NPIX * CQK;        // B*C*N bf16 = 8 MB

    proj_qk_kernel<<<BATCH * NPIX / 256, 256, 0, stream>>>(x, wq, bq, wk, bk, qbf, ktbf);
    proj_v_kernel<<<dim3(BATCH * NPIX / 256, 4), 256, 0, stream>>>(x, wv, bv, vtbf);
    attn_kernel<<<(BATCH * NPIX / 16) / 8, 256, 0, stream>>>(qbf, ktbf, vtbf, x, gamma, out);
}
